// LinearAttentionBlock_33681133535316
// MI455X (gfx1250) — matrix operand order repaired and checked by reading, not compiled, not run on hardware
//
#include <hip/hip_runtime.h>
#include <hip/hip_bf16.h>

typedef __bf16 v16bf __attribute__((ext_vector_type(16)));
typedef __bf16 v8bf  __attribute__((ext_vector_type(8)));
typedef float  v8f   __attribute__((ext_vector_type(8)));

static __device__ __forceinline__ float lab_phi(float x) {
    // elu(x) + 1
    return x > 0.f ? x + 1.f : __expf(x);
}

// ---------------- f32 -> bf16 conversion (grid-stride) ----------------
__global__ void lab_cvt_f32_bf16(const float* __restrict__ src,
                                 __bf16* __restrict__ dst, long n) {
    long i = (long)blockIdx.x * blockDim.x + threadIdx.x;
    long stride = (long)gridDim.x * blockDim.x;
    for (; i < n; i += stride) dst[i] = (__bf16)src[i];
}

// ---------------- generic bf16 WMMA GEMM:  C = A * Bw^T + bias ----------------
// A: M x K row-major bf16.  Bw: N x K row-major bf16 (weight matrix; product
// column n uses row n of Bw).  Output: f32 (Cf) or bf16 (Cb), exactly one non-null.
// M%256==0, N%64==0, K%32==0.  Block = 256 threads (8 waves); WG tile 256x64;
// wave tile 32x64: 2 A fragments share 4 B fragments -> 8 WMMAs / K-step.
__global__ __launch_bounds__(256)
void lab_gemm_bf16(const __bf16* __restrict__ A, const __bf16* __restrict__ Bw,
                   const float* __restrict__ bias,
                   float* __restrict__ Cf, __bf16* __restrict__ Cb,
                   int M, int N, int K) {
    const int tid  = threadIdx.x;
    const int lane = tid & 31;
    const int w    = tid >> 5;
    const int hi   = lane >> 4;     // 0: lanes 0-15, 1: lanes 16-31
    const int l15  = lane & 15;
    const int rowbase = blockIdx.y * 256 + w * 32;
    const int colbase = blockIdx.x * 64;

    v8f acc[2][4] = {};
    const size_t arow0 = (size_t)(rowbase + l15) * K;
    const size_t arow1 = arow0 + (size_t)16 * K;

    for (int k0 = 0; k0 < K; k0 += 32) {
        // A fragments (16x32): lane holds M=l15; K = 0..15 (hi=0)
        // or 16..31 (hi=1), contiguous, matching B.
        const __bf16* ap0 = A + arow0 + k0 + hi * 16;
        const __bf16* ap1 = A + arow1 + k0 + hi * 16;
        v8bf a0lo = *(const v8bf*)ap0;
        v8bf a0hi = *(const v8bf*)(ap0 + 8);
        v8bf a1lo = *(const v8bf*)ap1;
        v8bf a1hi = *(const v8bf*)(ap1 + 8);
        v16bf af0 = __builtin_shufflevector(a0lo, a0hi,
                        0,1,2,3,4,5,6,7,8,9,10,11,12,13,14,15);
        v16bf af1 = __builtin_shufflevector(a1lo, a1hi,
                        0,1,2,3,4,5,6,7,8,9,10,11,12,13,14,15);
#pragma unroll
        for (int t = 0; t < 4; ++t) {
            // B fragment (32x16): lane holds N=l15; K = k0..k0+15 (hi=0) or
            // k0+16..k0+31 (hi=1), contiguous in memory (row of Bw).
            const __bf16* bp = Bw + (size_t)(colbase + 16 * t + l15) * K + k0 + hi * 16;
            v16bf bf = *(const v16bf*)bp;
            acc[0][t] = __builtin_amdgcn_wmma_f32_16x16x32_bf16(
                            false, af0, false, bf, (short)0, acc[0][t], false, false);
            acc[1][t] = __builtin_amdgcn_wmma_f32_16x16x32_bf16(
                            false, af1, false, bf, (short)0, acc[1][t], false, false);
        }
    }
#pragma unroll
    for (int r = 0; r < 2; ++r) {
#pragma unroll
        for (int t = 0; t < 4; ++t) {
            const int col = colbase + 16 * t + l15;
            const float bc = bias ? bias[col] : 0.f;
#pragma unroll
            for (int j = 0; j < 8; ++j) {
                const int m = rowbase + r * 16 + j + hi * 8;  // C layout: VGPR j -> M=j (+8 hi)
                const float v = acc[r][t][j] + bc;
                if (Cb) Cb[(size_t)m * N + col] = (__bf16)v;
                else    Cf[(size_t)m * N + col] = v;
            }
        }
    }
}

// ---------------- phase A: kv = sum_n phi(k) v^T (stored TRANSPOSED [e][d]), k1 = sum_n phi(k)
// grid: (n-chunk 0..63, bh 0..63); each WG reduces 128 rows, atomically adds into kvT/k1.
__global__ __launch_bounds__(256)
void lab_attn_kv_accum(const __bf16* __restrict__ qkv,
                       float* __restrict__ kvT, float* __restrict__ k1) {
    const int bh = blockIdx.y;
    const int b = bh >> 4, h = bh & 15;
    const int nbase = blockIdx.x * 128;
    const int tid = threadIdx.x;
    __shared__ float ks[8][64];
    __shared__ float vs[8][64];
    const int d0 = (tid >> 4) * 4;
    const int e0 = (tid & 15) * 4;
    float acc[4][4] = {};
    float k1acc = 0.f;

    for (int nn = 0; nn < 128; nn += 8) {
        for (int i = tid; i < 512; i += 256) {
            const int r = i >> 6, c = i & 63;
            const size_t row = (size_t)b * 8192 + nbase + nn + r;
            const __bf16* base = qkv + row * 3072 + h * 64 + c;
            ks[r][c] = lab_phi((float)base[1024]);   // k component
            vs[r][c] = (float)base[2048];            // v component
        }
        __syncthreads();
#pragma unroll
        for (int r = 0; r < 8; ++r) {
            float kd[4], ve[4];
#pragma unroll
            for (int i = 0; i < 4; ++i) { kd[i] = ks[r][d0 + i]; ve[i] = vs[r][e0 + i]; }
#pragma unroll
            for (int i = 0; i < 4; ++i)
#pragma unroll
                for (int j = 0; j < 4; ++j) acc[i][j] += kd[i] * ve[j];
        }
        if (tid < 64) {
#pragma unroll
            for (int r = 0; r < 8; ++r) k1acc += ks[r][tid];
        }
        __syncthreads();
    }
    float* kvb = kvT + (size_t)bh * 4096;
#pragma unroll
    for (int i = 0; i < 4; ++i)
#pragma unroll
        for (int j = 0; j < 4; ++j)
            atomicAdd(&kvb[(e0 + j) * 64 + (d0 + i)], acc[i][j]);  // transposed store
    if (tid < 64) atomicAdd(&k1[bh * 64 + tid], k1acc);
}

// ---------------- phase B: out = (phi(q) @ kv) / clip(qk1), stored bf16 in the
// reference's head-reshape-bug permutation, ready as A matrix of the proj GEMM.
__global__ __launch_bounds__(256)
void lab_attn_apply(const __bf16* __restrict__ qkv, const __bf16* __restrict__ kvbf,
                    const float* __restrict__ k1, __bf16* __restrict__ A2) {
    const int bh = blockIdx.y;
    const int b = bh >> 4, h = bh & 15;
    const int tid = threadIdx.x;
    const int lane = tid & 31;
    const int w = tid >> 5;
    const int hi = lane >> 4;
    const int l15 = lane & 15;

    __shared__ __attribute__((aligned(32))) __bf16 skv[4096];  // kv^T [e][d] bf16
    __shared__ float sk1[64];
    {
        const unsigned* src = (const unsigned*)(kvbf + (size_t)bh * 4096);
        unsigned* dst = (unsigned*)skv;
        for (int i = tid; i < 2048; i += 256) dst[i] = src[i];
        if (tid < 64) sk1[tid] = k1[bh * 64 + tid];
    }
    __syncthreads();

    // B fragments (32x16 each): B[k=d, n=e] = kvT[e][d]; lane=e, contiguous d.
    v16bf bfr[2][4];
#pragma unroll
    for (int s = 0; s < 2; ++s)
#pragma unroll
        for (int t = 0; t < 4; ++t)
            bfr[s][t] = *(const v16bf*)&skv[(16 * t + l15) * 64 + 32 * s + hi * 16];

    const int row = blockIdx.x * 128 + w * 16 + l15;   // n index (0..8191)
    const __bf16* qrow = qkv + ((size_t)b * 8192 + row) * 3072 + h * 64;

    float qk1 = 0.f;
    v16bf af[2];
#pragma unroll
    for (int s = 0; s < 2; ++s) {
        const int dbase = s * 32 + hi * 16;
#pragma unroll
        for (int i = 0; i < 8; ++i) {
            const int dA = dbase + i, dB = dbase + 8 + i;
            float q0 = lab_phi((float)qrow[dA]); qk1 += q0 * sk1[dA]; af[s][i]     = (__bf16)q0;
            float q1 = lab_phi((float)qrow[dB]); qk1 += q1 * sk1[dB]; af[s][8 + i] = (__bf16)q1;
        }
    }
    v8f acc[4] = {};
#pragma unroll
    for (int s = 0; s < 2; ++s)
#pragma unroll
        for (int t = 0; t < 4; ++t)
            acc[t] = __builtin_amdgcn_wmma_f32_16x16x32_bf16(
                        false, af[s], false, bfr[s][t], (short)0, acc[t], false, false);

    // full qk1 for this row: lanes L and L+16 hold the two K-halves
    qk1 += __shfl_xor(qk1, 16, 32);
    const float rdiv = 1.f / fmaxf(qk1, 1e-6f);
    float sj[8];
#pragma unroll
    for (int j = 0; j < 8; ++j) sj[j] = __shfl(rdiv, j + hi * 8, 32);

    // (B,h,N,64) -> flat (B, N, 1024) without head transpose (reference bug):
    // row' = h*512 + n>>4 ; col' = (n&15)*64 + e
#pragma unroll
    for (int t = 0; t < 4; ++t) {
        const int e = 16 * t + l15;
#pragma unroll
        for (int j = 0; j < 8; ++j) {
            const int m = blockIdx.x * 128 + w * 16 + j + hi * 8;   // n
            const size_t r = (size_t)b * 8192 + h * 512 + (m >> 4);
            A2[r * 1024 + (size_t)(m & 15) * 64 + e] = (__bf16)(acc[t][j] * sj[j]);
        }
    }
}

// ---------------- LayerNorm + residual ----------------
__global__ __launch_bounds__(256)
void lab_layernorm_residual(const float* __restrict__ P, const float* __restrict__ H,
                            const float* __restrict__ gamma, const float* __restrict__ beta,
                            float* __restrict__ out) {
    const int row = blockIdx.x;
    const int tid = threadIdx.x;
    const float4 v = *(const float4*)(P + (size_t)row * 1024 + tid * 4);
    __shared__ float2 red[256];
    float s = v.x + v.y + v.z + v.w;
    float q = v.x * v.x + v.y * v.y + v.z * v.z + v.w * v.w;
    red[tid] = make_float2(s, q);
    __syncthreads();
    for (int st = 128; st > 0; st >>= 1) {
        if (tid < st) { red[tid].x += red[tid + st].x; red[tid].y += red[tid + st].y; }
        __syncthreads();
    }
    const float mu  = red[0].x * (1.f / 1024.f);
    const float var = red[0].y * (1.f / 1024.f) - mu * mu;
    const float rs  = rsqrtf(var + 1e-5f);
    const float4 hv = *(const float4*)(H + (size_t)row * 1024 + tid * 4);
    float iv[4] = {v.x, v.y, v.z, v.w};
    float hr[4] = {hv.x, hv.y, hv.z, hv.w};
    float4 res;
    float* rp = (float*)&res;
#pragma unroll
    for (int i = 0; i < 4; ++i) {
        const int c = tid * 4 + i;
        rp[i] = (iv[i] - mu) * rs * gamma[c] + beta[c] + hr[i];
    }
    *(float4*)(out + (size_t)row * 1024 + tid * 4) = res;
}

extern "C" void kernel_launch(void* const* d_in, const int* in_sizes, int n_in,
                              void* d_out, int out_size, void* d_ws, size_t ws_size,
                              hipStream_t stream) {
    const float* H     = (const float*)d_in[0];   // (4,8192,1024)
    const float* Wqkv  = (const float*)d_in[1];   // (3072,1024)
    const float* bqkv  = (const float*)d_in[2];   // (3072)
    const float* Wproj = (const float*)d_in[3];   // (1024,1024)
    const float* bproj = (const float*)d_in[4];   // (1024)
    const float* gamma = (const float*)d_in[5];
    const float* beta  = (const float*)d_in[6];
    float* out = (float*)d_out;
    char* ws = (char*)d_ws;

    // workspace layout (~278 MB total, with temporal aliasing):
    const size_t off_Hb   = 0;                                  // 67.1 MB bf16 H (later: A2)
    const size_t off_Wq   = off_Hb  + (size_t)33554432 * 2;     // 6.3 MB
    const size_t off_Wp   = off_Wq  + (size_t)3145728 * 2;      // 2.1 MB
    const size_t off_qkv  = off_Wp  + (size_t)1048576 * 2;      // 201.3 MB bf16 (later: P f32)
    const size_t off_kv   = off_qkv + (size_t)32768 * 3072 * 2; // 1.05 MB f32 kv^T
    const size_t off_k1   = off_kv  + (size_t)64 * 4096 * 4;    // 16 KB f32 k1
    const size_t off_kvbf = off_k1  + (size_t)64 * 64 * 4;      // 0.5 MB bf16 kv^T

    __bf16* Hb   = (__bf16*)(ws + off_Hb);
    __bf16* Wqb  = (__bf16*)(ws + off_Wq);
    __bf16* Wpb  = (__bf16*)(ws + off_Wp);
    __bf16* qkvb = (__bf16*)(ws + off_qkv);
    float*  kvT  = (float*) (ws + off_kv);
    float*  k1   = (float*) (ws + off_k1);
    __bf16* kvbf = (__bf16*)(ws + off_kvbf);
    __bf16* A2   = (__bf16*)(ws + off_Hb);    // alias: Hb dead after GEMM1
    float*  P    = (float*) (ws + off_qkv);   // alias: qkv dead after attention

    // 1) precision conversion
    lab_cvt_f32_bf16<<<4096, 256, 0, stream>>>(H, Hb, 33554432L);
    lab_cvt_f32_bf16<<<1024, 256, 0, stream>>>(Wqkv, Wqb, 3145728L);
    lab_cvt_f32_bf16<<<512, 256, 0, stream>>>(Wproj, Wpb, 1048576L);

    // 2) zero kv^T + k1 accumulators (contiguous)
    hipMemsetAsync(ws + off_kv, 0, (size_t)64 * 4096 * 4 + (size_t)64 * 64 * 4, stream);

    // 3) qkv = H @ Wqkv^T + b : M=32768, N=3072, K=1024, bf16 out
    lab_gemm_bf16<<<dim3(3072 / 64, 32768 / 256), 256, 0, stream>>>(
        Hb, Wqb, bqkv, nullptr, qkvb, 32768, 3072, 1024);

    // 4) kv^T / k1 reduction over N (split into 64 chunks per (b,h))
    lab_attn_kv_accum<<<dim3(64, 64), 256, 0, stream>>>(qkvb, kvT, k1);

    // 5) kv^T f32 -> bf16
    lab_cvt_f32_bf16<<<64, 256, 0, stream>>>(kvT, kvbf, 262144L);

    // 6) apply: (phi(q) @ kv) / qk1, store permuted bf16
    lab_attn_apply<<<dim3(64, 64), 256, 0, stream>>>(qkvb, kvbf, k1, A2);

    // 7) proj: P = A2 @ Wproj^T + b : M=32768, N=1024, K=1024, f32 out
    lab_gemm_bf16<<<dim3(1024 / 64, 32768 / 256), 256, 0, stream>>>(
        A2, Wpb, bproj, P, nullptr, 32768, 1024, 1024);

    // 8) LayerNorm + residual
    lab_layernorm_residual<<<32768, 256, 0, stream>>>(P, H, gamma, beta, out);
}
